// SelfAttention_67216238183127
// MI455X (gfx1250) — matrix operand order repaired and checked by reading, not compiled, not run on hardware
//
#include <hip/hip_runtime.h>
#include <hip/hip_bf16.h>

// ---------------------------------------------------------------------------
// Self-attention for MI455X (gfx1250, wave32, WMMA bf16 -> f32 accumulate)
// B=4, S=2048, D=1024, H=16, hd=64
// ---------------------------------------------------------------------------

typedef __attribute__((ext_vector_type(16))) __bf16 v16bf;
typedef __attribute__((ext_vector_type(8)))  __bf16 bf16x8;
typedef __attribute__((ext_vector_type(8)))  float  v8f;

static constexpr int BB = 4, SS = 2048, DD = 1024, HH = 16, HD = 64;
static constexpr int MROWS = BB * SS;   // 8192
static constexpr int NQKV  = 3 * DD;    // 3072

// ---- fragment loaders (layouts per CDNA5 ISA 7.12.2) ----------------------
// A fragment: 16x32 bf16. lane<16: row=lane, K={0..7,16..23}; lane>=16: K={8..15,24..31}
__device__ inline v16bf load_a_frag(const __bf16* base, int lda, int lane) {
  const int l = lane & 15, hi = lane >> 4;
  const __bf16* row = base + (size_t)l * lda + hi * 8;
  bf16x8 lo = *reinterpret_cast<const bf16x8*>(row);
  bf16x8 h8 = *reinterpret_cast<const bf16x8*>(row + 16);
  v16bf a;
#pragma unroll
  for (int i = 0; i < 8; ++i) { a[i] = lo[i]; a[i + 8] = h8[i]; }
  return a;
}

// B fragment: 32x16 bf16 from a K-contiguous [N][K] source (Bt[n*ldb + k]).
__device__ inline v16bf load_bt_frag(const __bf16* base, int ldb, int lane) {
  const int n = lane & 15, hi = lane >> 4;
  const __bf16* row = base + (size_t)n * ldb + hi * 8;
  bf16x8 lo = *reinterpret_cast<const bf16x8*>(row);
  bf16x8 h8 = *reinterpret_cast<const bf16x8*>(row + 16);
  v16bf b;
#pragma unroll
  for (int i = 0; i < 8; ++i) { b[i] = lo[i]; b[i + 8] = h8[i]; }
  return b;
}

__device__ inline v8f wmma_bf16(v16bf a, v16bf b, v8f c) {
  return __builtin_amdgcn_wmma_f32_16x16x32_bf16(false, a, false, b, (short)0, c,
                                                 false, false);
}

// ---- precision conversion / weight transpose ------------------------------
__global__ void cvt_f32_bf16(const float* __restrict__ in, __bf16* __restrict__ out,
                             int n) {
  int i = blockIdx.x * blockDim.x + threadIdx.x;
  int stride = gridDim.x * blockDim.x;
  for (; i < n; i += stride) out[i] = (__bf16)in[i];
}

// out[n*rowsK + k] = (bf16) in[k*colsN + n]
__global__ void transpose_bf16(const float* __restrict__ in, __bf16* __restrict__ out,
                               int rowsK, int colsN) {
  int i = blockIdx.x * blockDim.x + threadIdx.x;
  int total = rowsK * colsN;
  int stride = gridDim.x * blockDim.x;
  for (; i < total; i += stride) {
    int n = i / rowsK, k = i - n * rowsK;
    out[i] = (__bf16)in[(size_t)k * colsN + n];
  }
}

// ---------------------------------------------------------------------------
// GEMM core: wave tile 32(M) x 64(N), double-buffered fragments.
// Block = 8 waves (2Mx4N) -> 64 x 256 block tile.
// ---------------------------------------------------------------------------
struct GemmAcc {
  v8f acc[2][4];
};

__device__ inline void gemm_core(const __bf16* __restrict__ aPtr, int lda,
                                 const __bf16* __restrict__ bPtr, int ldb,
                                 int kdim, int lane, GemmAcc& g) {
#pragma unroll
  for (int i = 0; i < 2; ++i)
#pragma unroll
    for (int j = 0; j < 4; ++j) g.acc[i][j] = v8f{};

  v16bf a0[2], a1[2], b0[4], b1[4];

  auto loadA = [&](v16bf (&af)[2], int k) {
#pragma unroll
    for (int i = 0; i < 2; ++i)
      af[i] = load_a_frag(aPtr + (size_t)(i * 16) * lda + k, lda, lane);
  };
  auto loadB = [&](v16bf (&bf)[4], int k) {
#pragma unroll
    for (int j = 0; j < 4; ++j)
      bf[j] = load_bt_frag(bPtr + (size_t)(j * 16) * ldb + k, ldb, lane);
  };
  auto compute = [&](v16bf (&af)[2], v16bf (&bf)[4]) {
#pragma unroll
    for (int i = 0; i < 2; ++i)
#pragma unroll
      for (int j = 0; j < 4; ++j)
        g.acc[i][j] = wmma_bf16(af[i], bf[j], g.acc[i][j]);
  };

  loadA(a0, 0);
  loadB(b0, 0);
  // kdim is a multiple of 64: process two 32-wide steps per trip, ping-pong.
  for (int k = 0; k < kdim; k += 64) {
    loadA(a1, k + 32);           // in flight while buffer-0 WMMAs run
    loadB(b1, k + 32);
    compute(a0, b0);
    if (k + 64 < kdim) {
      loadA(a0, k + 64);
      loadB(b0, k + 64);
    }
    compute(a1, b1);
  }
}

// ---- stage 1: qkv = x @ w_qkv + b_qkv  (bf16 out; V also stored transposed)
__global__ __launch_bounds__(256) void qkv_gemm(const __bf16* __restrict__ xb,
                                                const __bf16* __restrict__ wT,
                                                const float* __restrict__ bias,
                                                __bf16* __restrict__ qkv,
                                                __bf16* __restrict__ vt) {
  const int lane = threadIdx.x & 31, wave = threadIdx.x >> 5;
  const int mBase = blockIdx.x * 64 + (wave >> 2) * 32;
  const int nBase = blockIdx.y * 256 + (wave & 3) * 64;
  GemmAcc g;
  gemm_core(xb + (size_t)mBase * DD, DD, wT + (size_t)nBase * DD, DD, DD, lane, g);

  const int l = lane & 15, hi = lane >> 4;
#pragma unroll
  for (int i = 0; i < 2; ++i) {
#pragma unroll
    for (int j = 0; j < 4; ++j) {
#pragma unroll
      for (int r = 0; r < 8; ++r) {
        const int m = mBase + i * 16 + r + 8 * hi;
        const int n = nBase + j * 16 + l;
        const float v = g.acc[i][j][r] + bias[n];
        qkv[(size_t)m * NQKV + n] = (__bf16)v;
        if (n >= 2 * DD) {                     // V -> transposed per-(b,h) [64][S]
          const int dh = n - 2 * DD;
          const int h = dh >> 6, d = dh & 63;
          const int bb = m / SS, srow = m - bb * SS;
          vt[(((size_t)(bb * HH + h)) * HD + d) * SS + srow] = (__bf16)v;
        }
      }
    }
  }
}

// ---- stage 3: out = attn @ w_out + b_out (f32 output) ---------------------
__global__ __launch_bounds__(256) void out_gemm(const __bf16* __restrict__ ab,
                                                const __bf16* __restrict__ wT,
                                                const float* __restrict__ bias,
                                                float* __restrict__ out) {
  const int lane = threadIdx.x & 31, wave = threadIdx.x >> 5;
  const int mBase = blockIdx.x * 64 + (wave >> 2) * 32;
  const int nBase = blockIdx.y * 256 + (wave & 3) * 64;
  GemmAcc g;
  gemm_core(ab + (size_t)mBase * DD, DD, wT + (size_t)nBase * DD, DD, DD, lane, g);

  const int l = lane & 15, hi = lane >> 4;
#pragma unroll
  for (int i = 0; i < 2; ++i) {
#pragma unroll
    for (int j = 0; j < 4; ++j) {
#pragma unroll
      for (int r = 0; r < 8; ++r) {
        const int m = mBase + i * 16 + r + 8 * hi;
        const int n = nBase + j * 16 + l;
        out[(size_t)m * DD + n] = g.acc[i][j][r] + bias[n];
      }
    }
  }
}

// ---- stage 2: flash attention per (b,h) -----------------------------------
// block = 128 threads (4 waves); each wave owns 16 query rows -> 64-row Q tile
__global__ __launch_bounds__(128) void attention(const __bf16* __restrict__ qkv,
                                                 const __bf16* __restrict__ vt,
                                                 __bf16* __restrict__ aout) {
  __shared__ __attribute__((aligned(16))) __bf16 plds[4][16][40];
  const int lane = threadIdx.x & 31, wave = threadIdx.x >> 5;
  const int l = lane & 15, hi = lane >> 4;
  const int b = blockIdx.y >> 4, h = blockIdx.y & 15;
  const int qBase = blockIdx.x * 64 + wave * 16;

  const size_t batOff = (size_t)(b * SS) * NQKV;
  const __bf16* qPtr = qkv + batOff + (size_t)qBase * NQKV + h * HD;
  const __bf16* kPtr = qkv + batOff + DD + h * HD;
  const __bf16* vPtr = vt + ((size_t)(b * HH + h) * HD) * SS;

  const float scale = 0.125f;  // hd^-0.5 = 1/8
  v16bf qf[2];
#pragma unroll
  for (int c = 0; c < 2; ++c) {
    qf[c] = load_a_frag(qPtr + c * 32, NQKV, lane);
#pragma unroll
    for (int i = 0; i < 16; ++i) qf[c][i] = (__bf16)((float)qf[c][i] * scale);
  }

  v8f o[4] = {};
  float mrow[8], lrow[8];
#pragma unroll
  for (int r = 0; r < 8; ++r) { mrow[r] = -1e30f; lrow[r] = 0.f; }

  // K fragments for a 32-key tile: [key-group g][k-chunk c]
  auto loadK = [&](v16bf (&kb)[4], int j0) {
#pragma unroll
    for (int g = 0; g < 2; ++g)
#pragma unroll
      for (int c = 0; c < 2; ++c)
        kb[g * 2 + c] =
            load_bt_frag(kPtr + (size_t)(j0 + g * 16) * NQKV + c * 32, NQKV, lane);
  };

  auto step = [&](v16bf (&kb)[4], int j0) {
    // S tile: 16 q-rows x 32 keys
    v8f s0 = {}, s1 = {};
    s0 = wmma_bf16(qf[0], kb[0], s0);
    s0 = wmma_bf16(qf[1], kb[1], s0);
    s1 = wmma_bf16(qf[0], kb[2], s1);
    s1 = wmma_bf16(qf[1], kb[3], s1);

    // V fragments issued early: latency overlaps the softmax VALU work below
    v16bf vb[4];
#pragma unroll
    for (int ds = 0; ds < 4; ++ds)
      vb[ds] = load_bt_frag(vPtr + (size_t)(ds * 16) * SS + j0, SS, lane);

    // online softmax; row M = r + 8*hi lives across the 16 lanes of each half
    float alpha[8];
#pragma unroll
    for (int r = 0; r < 8; ++r) {
      float rmax = fmaxf(s0[r], s1[r]);
#pragma unroll
      for (int off = 1; off < 16; off <<= 1) rmax = fmaxf(rmax, __shfl_xor(rmax, off));
      const float mnew = fmaxf(mrow[r], rmax);
      const float p0 = __expf(s0[r] - mnew);
      const float p1 = __expf(s1[r] - mnew);
      float rsum = p0 + p1;
#pragma unroll
      for (int off = 1; off < 16; off <<= 1) rsum += __shfl_xor(rsum, off);
      alpha[r] = __expf(mrow[r] - mnew);
      lrow[r] = lrow[r] * alpha[r] + rsum;
      mrow[r] = mnew;
      s0[r] = p0; s1[r] = p1;
    }
    // P: C-layout (col-per-lane) -> A-layout (row-per-lane) via LDS
#pragma unroll
    for (int r = 0; r < 8; ++r) {
      plds[wave][r + 8 * hi][l]      = (__bf16)s0[r];
      plds[wave][r + 8 * hi][16 + l] = (__bf16)s1[r];
    }
    __syncthreads();
    v16bf pf = load_a_frag(&plds[wave][0][0], 40, lane);
    __syncthreads();
    // O = diag(alpha)*O + P @ V
#pragma unroll
    for (int ds = 0; ds < 4; ++ds) {
#pragma unroll
      for (int r = 0; r < 8; ++r) o[ds][r] *= alpha[r];
      o[ds] = wmma_bf16(pf, vb[ds], o[ds]);
    }
  };

  // double-buffered K tiles, 64 keys per trip (SS % 64 == 0)
  v16bf kbuf0[4], kbuf1[4];
  loadK(kbuf0, 0);
  for (int j0 = 0; j0 < SS; j0 += 64) {
    loadK(kbuf1, j0 + 32);
    step(kbuf0, j0);
    if (j0 + 64 < SS) loadK(kbuf0, j0 + 64);
    step(kbuf1, j0 + 32);
  }

  // normalize, store bf16 back in [B*S, D] for the output projection
#pragma unroll
  for (int r = 0; r < 8; ++r) lrow[r] = 1.f / lrow[r];
#pragma unroll
  for (int ds = 0; ds < 4; ++ds) {
#pragma unroll
    for (int r = 0; r < 8; ++r) {
      const int m = b * SS + qBase + r + 8 * hi;
      const int col = h * HD + ds * 16 + l;
      aout[(size_t)m * DD + col] = (__bf16)(o[ds][r] * lrow[r]);
    }
  }
}

// ---------------------------------------------------------------------------
extern "C" void kernel_launch(void* const* d_in, const int* in_sizes, int n_in,
                              void* d_out, int out_size, void* d_ws, size_t ws_size,
                              hipStream_t stream) {
  (void)in_sizes; (void)n_in; (void)out_size; (void)ws_size;
  const float* x     = (const float*)d_in[0];   // [4,2048,1024]
  const float* w_qkv = (const float*)d_in[1];   // [1024,3072]
  const float* b_qkv = (const float*)d_in[2];   // [3072]
  const float* w_out = (const float*)d_in[3];   // [1024,1024]
  const float* b_out = (const float*)d_in[4];   // [1024]
  float* out = (float*)d_out;                   // [4,2048,1024]

  char* ws = (char*)d_ws;
  __bf16* xb    = (__bf16*)(ws);                        // 16 MB  x in bf16
  __bf16* wqkvT = (__bf16*)(ws + 16777216);             //  6 MB  w_qkv^T bf16
  __bf16* woutT = (__bf16*)(ws + 23068672);             //  2 MB  w_out^T bf16
  __bf16* qkvb  = (__bf16*)(ws + 25165824);             // 48 MB  qkv bf16
  __bf16* vtb   = (__bf16*)(ws + 75497472);             // 16 MB  V^T per (b,h)
  __bf16* aoutb = (__bf16*)(ws + 92274688);             // 16 MB  attn out bf16

  cvt_f32_bf16<<<8192, 256, 0, stream>>>(x, xb, MROWS * DD);
  transpose_bf16<<<12288, 256, 0, stream>>>(w_qkv, wqkvT, DD, NQKV);
  transpose_bf16<<<4096, 256, 0, stream>>>(w_out, woutT, DD, DD);

  qkv_gemm<<<dim3(MROWS / 64, NQKV / 256), 256, 0, stream>>>(xb, wqkvT, b_qkv,
                                                             qkvb, vtb);
  attention<<<dim3(SS / 64, BB * HH), 128, 0, stream>>>(qkvb, vtb, aoutb);
  out_gemm<<<dim3(MROWS / 64, DD / 256), 256, 0, stream>>>(aoutb, woutT, b_out, out);
}
